// EvoMesh_20718922236071
// MI455X (gfx1250) — matrix operand order repaired and checked by reading, not compiled, not run on hardware
//
#include <hip/hip_runtime.h>
#include <math.h>

// ---------------------------------------------------------------------------
// CDNA5 / gfx1250 types
// ---------------------------------------------------------------------------
typedef __attribute__((ext_vector_type(16))) __bf16 v16bf;
typedef __attribute__((ext_vector_type(8)))  __bf16 v8bf;
typedef __attribute__((ext_vector_type(8)))  float  v8f;
typedef __attribute__((ext_vector_type(4)))  unsigned int v4u;
typedef __attribute__((ext_vector_type(4)))  int v4i;
typedef __attribute__((ext_vector_type(8)))  int v8i;

#define LATENT 128

#if defined(__has_builtin)
#if __has_builtin(__builtin_amdgcn_sched_barrier)
#define SCHED_BARRIER() __builtin_amdgcn_sched_barrier(0)
#endif
#if __has_builtin(__builtin_amdgcn_tensor_load_to_lds)
#define HAVE_TDM 1
#endif
#endif
#ifndef SCHED_BARRIER
#define SCHED_BARRIER()
#endif

// bf16 <-> f32 (round-to-nearest-even)
__device__ __forceinline__ unsigned short f2bf(float f) {
    unsigned u = __float_as_uint(f);
    u += 0x7FFFu + ((u >> 16) & 1u);
    return (unsigned short)(u >> 16);
}
__device__ __forceinline__ float bf2f(unsigned short h) {
    return __uint_as_float(((unsigned)h) << 16);
}

// Device-scope f32 atomic max via native CDNA5 instruction.
__device__ __forceinline__ void atomic_max_f32_dev(float* p, float v) {
    unsigned long long a = (unsigned long long)p;
    asm volatile("global_atomic_max_num_f32 %0, %1, off scope:SCOPE_DEV"
                 :: "v"(a), "v"(v) : "memory");
}

// ---------------------------------------------------------------------------
// Tensor Data Mover: 1-D block load global -> LDS (D# per ISA 8.3/8.4).
// ---------------------------------------------------------------------------
__device__ __forceinline__ void stage_weights(unsigned short* lds_dst,
                                              const unsigned short* gsrc,
                                              unsigned bytes, int tid) {
#ifdef HAVE_TDM
    if (tid < 32) {
        const unsigned long long ga = (unsigned long long)(const void*)gsrc;
        const unsigned lds_off = (unsigned)(unsigned long long)(void*)lds_dst;
        const unsigned n8 = bytes >> 3;              // 8-byte elements (<= 65535)
        v4u g0;
        g0.x = 1u;                                   // count=1 valid descriptor
        g0.y = lds_off;                              // lds_addr
        g0.z = (unsigned)ga;                         // global_addr[31:0]
        g0.w = (unsigned)((ga >> 32) & 0x01FFFFFFull) | 0x80000000u; // [56:32], type=2
        v8i g1;
        g1[0] = (int)(3u << 16);                     // data_size=8B; wg_mask=0
        g1[1] = (int)((n8 & 0xFFFFu) << 16);         // tensor_dim0[15:0]
        g1[2] = (int)(((n8 >> 16) & 0xFFFFu) | (1u << 16)); // dim0 hi | tensor_dim1=1
        g1[3] = (int)((n8 & 0xFFFFu) << 16);         // tile_dim0 = n8
        g1[4] = 1;                                   // tile_dim1=1, tile_dim2=0
        g1[5] = (int)n8;                             // tensor_dim0_stride lo
        g1[6] = 0;
        g1[7] = 0;
        const v4i z4 = {0, 0, 0, 0};
#if __clang_major__ >= 23
        const v8i z8 = {0, 0, 0, 0, 0, 0, 0, 0};
        __builtin_amdgcn_tensor_load_to_lds(g0, g1, z4, z4, z8, 0);
#else
        __builtin_amdgcn_tensor_load_to_lds(g0, g1, z4, z4, 0);
#endif
        __builtin_amdgcn_s_wait_tensorcnt(0);
    }
#else
    const uint4* src = (const uint4*)gsrc;
    uint4* dst = (uint4*)lds_dst;
    for (unsigned i = tid; i < bytes / 16; i += blockDim.x) dst[i] = src[i];
#endif
    __syncthreads();
}

// ---------------------------------------------------------------------------
// Per-tile epilogue: bias + activation + store (bf16 LDS tile or f32 tile).
// ---------------------------------------------------------------------------
template<int OSTR, bool RELU, bool F32OUT>
__device__ __forceinline__ void tile_epilogue(const v8f& acc, int nt, float bb,
    unsigned short* __restrict__ O, float* __restrict__ Of, int hsel, int nl)
{
    const int n = nt * 16 + nl;
#pragma unroll
    for (int r = 0; r < 8; ++r) {
        const int m = r + hsel * 8;
        float v = acc[r] + bb;
        if (RELU) v = fmaxf(v, 0.f);
        if (F32OUT) Of[m * OSTR + n] = v;
        else        O[m * OSTR + n]  = f2bf(v);
    }
}

// ---------------------------------------------------------------------------
// One wave: O[16 x 128] = act(A[16 x KA] * W + b), all tiles in LDS (bf16).
// Dual n-tile pipeline for KA<=256; sched_barrier keeps ds_loads ahead of
// the WMMA chain so waits are progressive instead of full stalls.
// ---------------------------------------------------------------------------
template<int KA, int ASTR, int WSTR, int OSTR, bool RELU, bool F32OUT>
__device__ __forceinline__ void wave_mlp16(
    const unsigned short* __restrict__ A,
    const unsigned short* __restrict__ Wt,
    const float* __restrict__ bias,
    unsigned short* __restrict__ O, float* __restrict__ Of, int lane)
{
    constexpr int KS = KA / 32;
    constexpr bool DUAL = (KS <= 8);
    const int hsel = lane >> 4;
    const int nl   = lane & 15;

    // bias preload (overlaps everything below)
    float bb[8];
#pragma unroll
    for (int nt = 0; nt < 8; ++nt) bb[nt] = bias[nt * 16 + nl];

    // preload all A fragments
    v16bf av[KS];
    {
        const unsigned short* arow = A + nl * ASTR + hsel * 16;
#pragma unroll
        for (int s = 0; s < KS; ++s) {
            v8bf lo = *(const v8bf*)(const void*)(arow + 32 * s);
            v8bf hi = *(const v8bf*)(const void*)(arow + 32 * s + 8);
            av[s] = __builtin_shufflevector(lo, hi,
                        0,1,2,3,4,5,6,7,8,9,10,11,12,13,14,15);
        }
    }

    if constexpr (DUAL) {
        for (int nt = 0; nt < 8; nt += 2) {
            const unsigned short* w0 = Wt + (nt * 16 + nl) * WSTR + hsel * 16;
            const unsigned short* w1 = w0 + 16 * WSTR;
            v16bf b0[KS], b1[KS];
#pragma unroll
            for (int s = 0; s < KS; ++s)
                b0[s] = *(const v16bf*)(const void*)(w0 + 32 * s);
#pragma unroll
            for (int s = 0; s < KS; ++s)
                b1[s] = *(const v16bf*)(const void*)(w1 + 32 * s);
            SCHED_BARRIER();
            v8f a0 = {}, a1 = {};
#pragma unroll
            for (int s = 0; s < KS; ++s) {
                a0 = __builtin_amdgcn_wmma_f32_16x16x32_bf16(
                         false, av[s], false, b0[s], (short)0, a0, false, false);
                a1 = __builtin_amdgcn_wmma_f32_16x16x32_bf16(
                         false, av[s], false, b1[s], (short)0, a1, false, false);
            }
            tile_epilogue<OSTR, RELU, F32OUT>(a0, nt,     bb[nt],     O, Of, hsel, nl);
            tile_epilogue<OSTR, RELU, F32OUT>(a1, nt + 1, bb[nt + 1], O, Of, hsel, nl);
        }
    } else {
        for (int nt = 0; nt < 8; ++nt) {
            const unsigned short* w0 = Wt + (nt * 16 + nl) * WSTR + hsel * 16;
            v16bf b0[KS];
#pragma unroll
            for (int s = 0; s < KS; ++s)
                b0[s] = *(const v16bf*)(const void*)(w0 + 32 * s);
            SCHED_BARRIER();
            v8f a0 = {};
#pragma unroll
            for (int s = 0; s < KS; ++s)
                a0 = __builtin_amdgcn_wmma_f32_16x16x32_bf16(
                         false, av[s], false, b0[s], (short)0, a0, false, false);
            tile_epilogue<OSTR, RELU, F32OUT>(a0, nt, bb[nt], O, Of, hsel, nl);
        }
    }
}

// ---------------------------------------------------------------------------
// Weight prep: f32 -> bf16, transposed [out][k], strides padded (see R1).
// ---------------------------------------------------------------------------
#define WBF_TOTAL 227328

__global__ void k_prep(const float* __restrict__ wei0, const float* __restrict__ wei1,
                       const float* __restrict__ wei2, const float* __restrict__ wew0,
                       const float* __restrict__ wew1, const float* __restrict__ wnd0,
                       const float* __restrict__ wnd1, const float* __restrict__ wnd2,
                       const float* __restrict__ wg0,  const float* __restrict__ wg1,
                       unsigned short* __restrict__ wbf)
{
    int t = blockIdx.x * blockDim.x + threadIdx.x;
    if (t >= WBF_TOTAL) return;
    int ofs, st, ks; const float* src;
    if      (t < 37888)  { ofs = 0;      st = 296; ks = 260; src = wei0; }
    else if (t < 55296)  { ofs = 37888;  st = 136; ks = 128; src = wei1; }
    else if (t < 72704)  { ofs = 55296;  st = 136; ks = 128; src = wei2; }
    else if (t < 90112)  { ofs = 72704;  st = 136; ks = 128; src = wew0; }
    else if (t < 107520) { ofs = 90112;  st = 136; ks = 128; src = wew1; }
    else if (t < 141312) { ofs = 107520; st = 264; ks = 256; src = wnd0; }
    else if (t < 158720) { ofs = 141312; st = 136; ks = 128; src = wnd1; }
    else if (t < 176128) { ofs = 158720; st = 136; ks = 128; src = wnd2; }
    else if (t < 209920) { ofs = 176128; st = 264; ks = 256; src = wg0;  }
    else                 { ofs = 209920; st = 136; ks = 128; src = wg1;  }
    const int local = t - ofs;
    const int n = local / st, k = local % st;
    wbf[t] = (k < ks) ? f2bf(src[k * 128 + n]) : (unsigned short)0;
}

// ---------------------------------------------------------------------------
// Edge kernel: 4 waves x 16 edges
// ---------------------------------------------------------------------------
#define EK_WT_ELEMS   107520
#define EK_WAVE_ELEMS 9088
#define EK_SMEM_BYTES (EK_WT_ELEMS * 2 + 4 * EK_WAVE_ELEMS * 2)   // 287744

__global__ __launch_bounds__(128)
__attribute__((amdgpu_waves_per_eu(1)))
void ek_edge_mlp(const float* __restrict__ x, const int* __restrict__ g,
                 const float* __restrict__ pos,
                 const unsigned short* __restrict__ wbf,
                 const float* __restrict__ b_ei0, const float* __restrict__ b_ei1,
                 const float* __restrict__ b_ei2,
                 const float* __restrict__ b_ew0, const float* __restrict__ b_ew1,
                 const float* __restrict__ w_ew2, const float* __restrict__ b_ew2,
                 unsigned short* __restrict__ eeBF, float* __restrict__ ewLogit,
                 int E)
{
    extern __shared__ unsigned short smem[];
    const int tid = threadIdx.x, lane = tid & 31, wave = tid >> 5;
    const int hsel = lane >> 4, nl = lane & 15;

    stage_weights(smem, wbf, EK_WT_ELEMS * 2, tid);   // TDM

    unsigned short* A0 = smem + EK_WT_ELEMS + wave * EK_WAVE_ELEMS; // 16x296
    unsigned short* hA = A0 + 4736;                                 // 16x136
    unsigned short* hB = hA + 2176;                                 // 16x136
    float* f32t = (float*)(void*)A0;                                // 16x132 overlay

    const int ebase = blockIdx.x * 64 + wave * 16;

    // fiber (d, |d|) + K padding
    if (lane < 16) {
        int e = ebase + lane;
        int ii = 0, jj = 0;
        if (e < E) { ii = g[e]; jj = g[E + e]; }
        float dx = pos[ii*3+0] - pos[jj*3+0];
        float dy = pos[ii*3+1] - pos[jj*3+1];
        float dz = pos[ii*3+2] - pos[jj*3+2];
        float nr = sqrtf(dx*dx + dy*dy + dz*dz);
        unsigned short* row = A0 + lane * 296;
        row[0] = f2bf(dx); row[1] = f2bf(dy); row[2] = f2bf(dz); row[3] = f2bf(nr);
        for (int k = 260; k < 296; ++k) row[k] = 0;
    }
    // gather x_i, x_j: one float4 per lane per row, vector LDS stores
    for (int r = 0; r < 16; ++r) {
        const int er = ebase + r;
        const bool vr = er < E;
        const int ir = vr ? g[er] : 0;
        const int jr = vr ? g[E + er] : 0;
        float4 xi = {0.f, 0.f, 0.f, 0.f}, xj = {0.f, 0.f, 0.f, 0.f};
        if (vr) {
            xi = *(const float4*)(x + (size_t)ir * 128 + lane * 4);
            xj = *(const float4*)(x + (size_t)jr * 128 + lane * 4);
        }
        ushort4 pi = { f2bf(xi.x), f2bf(xi.y), f2bf(xi.z), f2bf(xi.w) };
        ushort4 pj = { f2bf(xj.x), f2bf(xj.y), f2bf(xj.z), f2bf(xj.w) };
        *(ushort4*)(A0 + r*296 + 4   + lane*4) = pi;
        *(ushort4*)(A0 + r*296 + 132 + lane*4) = pj;
    }

    // edge_info: 260->128 relu, 128->128 relu, 128->128 (f32), LN
    wave_mlp16<288,296,296,136,true ,false>(A0, smem + 0,     b_ei0, hA, nullptr, lane);
    wave_mlp16<128,136,136,136,true ,false>(hA, smem + 37888, b_ei1, hB, nullptr, lane);
    wave_mlp16<128,136,136,132,false,true >(hB, smem + 55296, b_ei2, nullptr, f32t, lane);

    // LayerNorm: lane owns row nl, half hsel -> stats in-lane, two shuffles
    const float* frow = f32t + nl * 132 + hsel * 64;
    float sum = 0.f, sq = 0.f;
    for (int t2 = 0; t2 < 64; t2 += 4) {
        float4 v = *(const float4*)(frow + t2);
        sum += v.x + v.y + v.z + v.w;
        sq  += v.x*v.x + v.y*v.y + v.z*v.z + v.w*v.w;
    }
    sum += __shfl_xor(sum, 16, 32);
    sq  += __shfl_xor(sq,  16, 32);
    const float mean = sum * (1.f/128.f);
    const float scal = rsqrtf(fmaxf(sq * (1.f/128.f) - mean*mean, 0.f) + 1e-5f);
    {
        const int er = ebase + nl;
        unsigned short* hrow = hA + nl * 136 + hsel * 64;
        unsigned short* grow = eeBF + (size_t)(er < E ? er : 0) * 128 + hsel * 64;
        const bool vr = er < E;
        for (int t2 = 0; t2 < 64; t2 += 4) {
            float4 v = *(const float4*)(frow + t2);
            ushort4 pk = { f2bf((v.x - mean) * scal), f2bf((v.y - mean) * scal),
                           f2bf((v.z - mean) * scal), f2bf((v.w - mean) * scal) };
            *(ushort4*)(hrow + t2) = pk;
            if (vr) *(ushort4*)(grow + t2) = pk;
        }
    }

    // edge_weight: 128->128 relu, 128->128 relu, 128->1
    unsigned short* hC = A0;            // f32t dead
    wave_mlp16<128,136,136,136,true,false>(hA, smem + 72704, b_ew0, hC, nullptr, lane);
    wave_mlp16<128,136,136,136,true,false>(hC, smem + 90112, b_ew1, hB, nullptr, lane);
    {
        float part = 0.f;
        const unsigned short* hrow = hB + nl * 136 + hsel * 64;
        const float* wr = w_ew2 + hsel * 64;
        for (int t2 = 0; t2 < 64; t2 += 4) {
            ushort4 hv = *(const ushort4*)(hrow + t2);
            float4  wv = *(const float4*)(wr + t2);
            part += bf2f(hv.x)*wv.x + bf2f(hv.y)*wv.y
                  + bf2f(hv.z)*wv.z + bf2f(hv.w)*wv.w;
        }
        part += __shfl_xor(part, 16, 32);
        if (lane < 16) {
            const int er = ebase + nl;
            if (er < E) ewLogit[er] = part + b_ew2[0];
        }
    }
}

// ---------------------------------------------------------------------------
// Segment softmax + aggregation
// ---------------------------------------------------------------------------
__global__ void k_init(float* nmax, float* nsum, float* aggr, int N) {
    const int t = blockIdx.x * blockDim.x + threadIdx.x;
    if (t < N) { nmax[t] = -__builtin_inff(); nsum[t] = 0.f; }
    if (t < N * 128) aggr[t] = 0.f;
}

__global__ void k_max(const int* __restrict__ g, const float* __restrict__ ewLogit,
                      float* __restrict__ nmax, int E) {
    const int t = blockIdx.x * blockDim.x + threadIdx.x;
    if (t < E) atomic_max_f32_dev(&nmax[g[E + t]], ewLogit[t]);
}

__global__ void k_exp(const int* __restrict__ g, const float* __restrict__ ewLogit,
                      const float* __restrict__ nmax, float* __restrict__ eExp,
                      float* __restrict__ nsum, int E) {
    const int t = blockIdx.x * blockDim.x + threadIdx.x;
    if (t < E) {
        const int j = g[E + t];
        const float e = __expf(ewLogit[t] - nmax[j]);
        eExp[t] = e;
        atomicAdd(&nsum[j], e);
    }
}

__global__ void k_aggr(const int* __restrict__ g, const float* __restrict__ eExp,
                       const float* __restrict__ nsum,
                       const unsigned short* __restrict__ eeBF,
                       float* __restrict__ aggr, float* __restrict__ out_ew, int E) {
    const int t = blockIdx.x * blockDim.x + threadIdx.x;
    if (t < E * 128) {
        const int e = t >> 7, f = t & 127;
        const int j = g[E + e];
        const float w = eExp[e] / nsum[j];
        if (f == 0) out_ew[e] = w;
        atomicAdd(&aggr[(size_t)j * 128 + f], bf2f(eeBF[t]) * w);
    }
}

// ---------------------------------------------------------------------------
// Node kernel
// ---------------------------------------------------------------------------
#define NK_WT_ELEMS   119808
#define NK_WAVE_ELEMS 8576
#define NK_SMEM_BYTES (NK_WT_ELEMS * 2 + 4 * NK_WAVE_ELEMS * 2)   // 308224

__global__ __launch_bounds__(128)
__attribute__((amdgpu_waves_per_eu(1)))
void nk_node_mlp(const float* __restrict__ x, const float* __restrict__ aggr,
                 const float* __restrict__ gn, const float* __restrict__ temp,
                 const unsigned short* __restrict__ wbf,
                 const float* __restrict__ b_nd0, const float* __restrict__ b_nd1,
                 const float* __restrict__ b_nd2,
                 const float* __restrict__ b_g0, const float* __restrict__ b_g1,
                 const float* __restrict__ w_g2, const float* __restrict__ b_g2,
                 float* __restrict__ out_node, float* __restrict__ out_y, int N)
{
    extern __shared__ unsigned short smem[];
    const int tid = threadIdx.x, lane = tid & 31, wave = tid >> 5;
    const int hsel = lane >> 4, nl = lane & 15;

    stage_weights(smem, wbf, NK_WT_ELEMS * 2, tid);   // TDM

    unsigned short* A0 = smem + NK_WT_ELEMS + wave * NK_WAVE_ELEMS; // 16x264
    unsigned short* hA = A0 + 4224;
    unsigned short* hB = hA + 2176;
    float* f32t = (float*)(void*)A0;                                // 16x132

    const int nbase = blockIdx.x * 64 + wave * 16;

    // tmp2 = [x, aggr] : one float4 per lane per row
    for (int r = 0; r < 16; ++r) {
        const int nd = nbase + r;
        const bool vr = nd < N;
        const int ns = vr ? nd : 0;
        float4 xv = {0.f, 0.f, 0.f, 0.f}, av = {0.f, 0.f, 0.f, 0.f};
        if (vr) {
            xv = *(const float4*)(x    + (size_t)ns * 128 + lane * 4);
            av = *(const float4*)(aggr + (size_t)ns * 128 + lane * 4);
        }
        ushort4 px = { f2bf(xv.x), f2bf(xv.y), f2bf(xv.z), f2bf(xv.w) };
        ushort4 pa = { f2bf(av.x), f2bf(av.y), f2bf(av.z), f2bf(av.w) };
        *(ushort4*)(A0 + r*264 + lane*4)       = px;
        *(ushort4*)(A0 + r*264 + 128 + lane*4) = pa;
    }

    // gumbel L1 (A0 alive), node_delta L1, gumbel L2
    wave_mlp16<256,264,264,136,true,false>(A0, smem + 68608,  b_g0,  hB, nullptr, lane);
    wave_mlp16<256,264,264,136,true,false>(A0, smem + 0,      b_nd0, hA, nullptr, lane);
    unsigned short* gC = A0;
    wave_mlp16<128,136,136,136,true,false>(hB, smem + 102400, b_g1,  gC, nullptr, lane);

    // gumbel head: hard one-hot of argmax((logits+noise)/temp)
    {
        float p0 = 0.f, p1 = 0.f;
        const unsigned short* grow = gC + nl * 136 + hsel * 64;
        for (int t2 = 0; t2 < 64; t2 += 2) {
            const float4 wv = *(const float4*)(w_g2 + (hsel * 64 + t2) * 2);
            const float h0 = bf2f(grow[t2]), h1 = bf2f(grow[t2 + 1]);
            p0 += h0 * wv.x + h1 * wv.z;
            p1 += h0 * wv.y + h1 * wv.w;
        }
        p0 += __shfl_xor(p0, 16, 32);
        p1 += __shfl_xor(p1, 16, 32);
        if (lane < 16) {
            const int nd = nbase + nl;
            if (nd < N) {
                const float it = 1.f / temp[0];
                const float a0 = (p0 + b_g2[0] + gn[2*nd + 0]) * it;
                const float a1 = (p1 + b_g2[1] + gn[2*nd + 1]) * it;
                out_y[2*nd + 0] = (a1 > a0) ? 0.f : 1.f;
                out_y[2*nd + 1] = (a1 > a0) ? 1.f : 0.f;
            }
        }
    }

    // node_delta L2, L3 (f32), LN, +x
    wave_mlp16<128,136,136,136,true ,false>(hA, smem + 33792, b_nd1, hB, nullptr, lane);
    wave_mlp16<128,136,136,132,false,true >(hB, smem + 51200, b_nd2, nullptr, f32t, lane);

    const float* frow = f32t + nl * 132 + hsel * 64;
    float sum = 0.f, sq = 0.f;
    for (int t2 = 0; t2 < 64; t2 += 4) {
        float4 v = *(const float4*)(frow + t2);
        sum += v.x + v.y + v.z + v.w;
        sq  += v.x*v.x + v.y*v.y + v.z*v.z + v.w*v.w;
    }
    sum += __shfl_xor(sum, 16, 32);
    sq  += __shfl_xor(sq,  16, 32);
    const float mean = sum * (1.f/128.f);
    const float scal = rsqrtf(fmaxf(sq * (1.f/128.f) - mean*mean, 0.f) + 1e-5f);
    {
        const int nd = nbase + nl;
        if (nd < N) {
            const float* xr   = x        + (size_t)nd * 128 + hsel * 64;
            float*       orow = out_node + (size_t)nd * 128 + hsel * 64;
            for (int t2 = 0; t2 < 64; t2 += 4) {
                float4 v  = *(const float4*)(frow + t2);
                float4 xv = *(const float4*)(xr + t2);
                float4 o  = { (v.x - mean) * scal + xv.x, (v.y - mean) * scal + xv.y,
                              (v.z - mean) * scal + xv.z, (v.w - mean) * scal + xv.w };
                *(float4*)(orow + t2) = o;
            }
        }
    }
}

// ---------------------------------------------------------------------------
// Host side
// ---------------------------------------------------------------------------
extern "C" void kernel_launch(void* const* d_in, const int* in_sizes, int n_in,
                              void* d_out, int out_size, void* d_ws, size_t ws_size,
                              hipStream_t stream) {
    const float* x    = (const float*)d_in[0];
    const int*   g    = (const int*)d_in[1];
    const float* pos  = (const float*)d_in[2];
    const float* temp = (const float*)d_in[3];
    const float* gn   = (const float*)d_in[4];
    const int E = in_sizes[1] / 2;
    const int N = in_sizes[2] / 3;

    unsigned short* wbf = (unsigned short*)d_ws;
    float* wsF     = (float*)((char*)d_ws + (size_t)WBF_TOTAL * 2);
    float* ewLogit = wsF;
    float* eExp    = ewLogit + E;
    float* nmax    = eExp + E;
    float* nsum    = nmax + N;
    float* aggr    = nsum + N;
    unsigned short* eeBF = (unsigned short*)(aggr + (size_t)N * 128);

    float* out     = (float*)d_out;
    float* out_ew  = out + (size_t)N * 128;     // T==1
    float* out_y   = out_ew + E;

    (void)hipFuncSetAttribute((const void*)ek_edge_mlp,
        hipFuncAttributeMaxDynamicSharedMemorySize, EK_SMEM_BYTES);
    (void)hipFuncSetAttribute((const void*)nk_node_mlp,
        hipFuncAttributeMaxDynamicSharedMemorySize, NK_SMEM_BYTES);

    k_prep<<<(WBF_TOTAL + 255) / 256, 256, 0, stream>>>(
        (const float*)d_in[5], (const float*)d_in[7], (const float*)d_in[9],
        (const float*)d_in[11], (const float*)d_in[13],
        (const float*)d_in[23], (const float*)d_in[25], (const float*)d_in[27],
        (const float*)d_in[17], (const float*)d_in[19], wbf);

    ek_edge_mlp<<<(E + 63) / 64, 128, EK_SMEM_BYTES, stream>>>(
        x, g, pos, wbf,
        (const float*)d_in[6], (const float*)d_in[8], (const float*)d_in[10],
        (const float*)d_in[12], (const float*)d_in[14],
        (const float*)d_in[15], (const float*)d_in[16],
        eeBF, ewLogit, E);

    k_init<<<(N * 128 + 255) / 256, 256, 0, stream>>>(nmax, nsum, aggr, N);
    k_max<<<(E + 255) / 256, 256, 0, stream>>>(g, ewLogit, nmax, E);
    k_exp<<<(E + 255) / 256, 256, 0, stream>>>(g, ewLogit, nmax, eExp, nsum, E);

    const long long tot = (long long)E * 128;
    k_aggr<<<(int)((tot + 255) / 256), 256, 0, stream>>>(
        g, eExp, nsum, eeBF, aggr, out_ew, E);

    nk_node_mlp<<<(N + 63) / 64, 128, NK_SMEM_BYTES, stream>>>(
        x, aggr, gn, temp, wbf + 107520,
        (const float*)d_in[24], (const float*)d_in[26], (const float*)d_in[28],
        (const float*)d_in[18], (const float*)d_in[20],
        (const float*)d_in[21], (const float*)d_in[22],
        out, out_y, N);
}
